// PhysicsSSM_3882650437058
// MI455X (gfx1250) — compile-verified, matrix operand order checked by reading, not run on hardware
//
#include <hip/hip_runtime.h>
#include <hip/hip_bf16.h>

// ---------------------------------------------------------------------------
// Problem constants (PhysicsSSM reference): B=8, L=2048, D=256, N=32, P=64
// ---------------------------------------------------------------------------
#define BB   8
#define LL   2048
#define DD   256
#define NN   32
#define PP   64
#define DI   512            // 2*D
#define DTR  16
#define KCONV 4
#define MM   (BB*LL)        // 16384 rows
#define DBCN 128            // x_proj output padded 80 -> 128 (NT=8, mult of 4)

typedef __attribute__((ext_vector_type(16))) __bf16 v16bf;
typedef __attribute__((ext_vector_type(8)))  float  v8f;

// ---------------------------------------------------------------------------
// helpers
// ---------------------------------------------------------------------------
static __device__ __forceinline__ unsigned short f32_to_bf16(float f) {
    unsigned u = __float_as_uint(f);
    unsigned r = u + 0x7FFFu + ((u >> 16) & 1u);   // round-to-nearest-even
    return (unsigned short)(r >> 16);
}
static __device__ __forceinline__ float silu_f(float x) {
    return x / (1.f + __expf(-x));
}
static __device__ __forceinline__ float sigmoid_f(float x) {
    return 1.f / (1.f + __expf(-x));
}
static __device__ __forceinline__ v8f wmma_bf16(v16bf a, v16bf b, v8f c) {
    return __builtin_amdgcn_wmma_f32_16x16x32_bf16(false, a, false, b, (short)0, c, false, false);
}

// ---------------------------------------------------------------------------
// LayerNorm over D=256 (one block per row, 256 threads)
// ---------------------------------------------------------------------------
__global__ void layernorm_k(const float* __restrict__ x,
                            const float* __restrict__ w,
                            const float* __restrict__ b,
                            float* __restrict__ y) {
    __shared__ float s[8];
    int row = blockIdx.x, tid = threadIdx.x;
    float v = x[(size_t)row * DD + tid];
    float sum = v;
    #pragma unroll
    for (int o = 16; o >= 1; o >>= 1) sum += __shfl_xor(sum, o, 32);
    if ((tid & 31) == 0) s[tid >> 5] = sum;
    __syncthreads();
    float mu = 0.f;
    #pragma unroll
    for (int i = 0; i < 8; ++i) mu += s[i];
    mu *= (1.f / DD);
    __syncthreads();
    float d = v - mu;
    float sq = d * d;
    #pragma unroll
    for (int o = 16; o >= 1; o >>= 1) sq += __shfl_xor(sq, o, 32);
    if ((tid & 31) == 0) s[tid >> 5] = sq;
    __syncthreads();
    float var = 0.f;
    #pragma unroll
    for (int i = 0; i < 8; ++i) var += s[i];
    var *= (1.f / DD);
    y[(size_t)row * DD + tid] = d * __frsqrt_rn(var + 1e-5f) * w[tid] + b[tid];
}

// ---------------------------------------------------------------------------
// Pack fp32 row-major [M x stride] activations -> bf16 WMMA A-fragments.
// Fragment layout (ISA 7.12.2, 16-bit A 16x32): lane = g*16+m holds row m of
// the 16-row tile; element e maps to K-offset 16*(e>>3) + 8*g + (e&7).
// dst index: ((mt*KTtot + ktOff + kt)*32 + lane)*16 + e   (32B/lane)
// ---------------------------------------------------------------------------
__global__ void pack_a_k(const float* __restrict__ src, unsigned short* __restrict__ dst,
                         int MT, int KT, int stride, int ktOff, int KTtot) {
    int idx = blockIdx.x * blockDim.x + threadIdx.x;
    if (idx >= MT * KT * 32) return;
    int lane = idx & 31;
    int t    = idx >> 5;
    int kt   = t % KT;
    int mt   = t / KT;
    int g = lane >> 4, m = lane & 15;
    const float* srow = src + (size_t)(mt * 16 + m) * stride + kt * 32;
    unsigned short v[16];
    #pragma unroll
    for (int e = 0; e < 16; ++e) {
        int k = 16 * (e >> 3) + 8 * g + (e & 7);
        v[e] = f32_to_bf16(srow[k]);
    }
    size_t o = ((size_t)(mt * KTtot + ktOff + kt) * 32 + lane) * 16;
    uint4* d4 = (uint4*)(dst + o);
    uint4 w0, w1;
    w0.x = v[0]  | ((unsigned)v[1]  << 16); w0.y = v[2]  | ((unsigned)v[3]  << 16);
    w0.z = v[4]  | ((unsigned)v[5]  << 16); w0.w = v[6]  | ((unsigned)v[7]  << 16);
    w1.x = v[8]  | ((unsigned)v[9]  << 16); w1.y = v[10] | ((unsigned)v[11] << 16);
    w1.z = v[12] | ((unsigned)v[13] << 16); w1.w = v[14] | ((unsigned)v[15] << 16);
    d4[0] = w0; d4[1] = w1;
}

// Pack fp32 weights [K x Nlog] row-major -> bf16 WMMA B-fragments, zero-padded
// out to NT*16 columns. lane = g*16+n holds column n; same K-offset rule as A.
__global__ void pack_w_k(const float* __restrict__ src, unsigned short* __restrict__ dst,
                         int KT, int NT, int Nlog) {
    int idx = blockIdx.x * blockDim.x + threadIdx.x;
    if (idx >= KT * NT * 32) return;
    int lane = idx & 31;
    int t    = idx >> 5;
    int nt   = t % NT;
    int kt   = t / NT;
    int g = lane >> 4, n = lane & 15;
    int col = nt * 16 + n;
    unsigned short v[16];
    #pragma unroll
    for (int e = 0; e < 16; ++e) {
        int k = kt * 32 + 16 * (e >> 3) + 8 * g + (e & 7);
        v[e] = (col < Nlog) ? f32_to_bf16(src[(size_t)k * Nlog + col]) : (unsigned short)0;
    }
    size_t o = ((size_t)(kt * NT + nt) * 32 + lane) * 16;
    uint4* d4 = (uint4*)(dst + o);
    uint4 w0, w1;
    w0.x = v[0]  | ((unsigned)v[1]  << 16); w0.y = v[2]  | ((unsigned)v[3]  << 16);
    w0.z = v[4]  | ((unsigned)v[5]  << 16); w0.w = v[6]  | ((unsigned)v[7]  << 16);
    w1.x = v[8]  | ((unsigned)v[9]  << 16); w1.y = v[10] | ((unsigned)v[11] << 16);
    w1.z = v[12] | ((unsigned)v[13] << 16); w1.w = v[14] | ((unsigned)v[15] << 16);
    d4[0] = w0; d4[1] = w1;
}

// ---------------------------------------------------------------------------
// bf16 WMMA GEMM v2: C[M,N] = A*W + bias (+SiLU).
// Block = 256 threads = 8 waves; each wave owns 2 M-tiles (32 rows); the whole
// block shares one 4-N-tile strip (64 cols) -> 256x64 per block.
// B tiles are staged to LDS via global_load_async_to_lds_b128 (ASYNCcnt),
// double-buffered: issue batch kt+1, s_wait_asynccnt 0x1 leaves it in flight
// while batch kt is consumed. A fragments are register-pipelined.
// __launch_bounds__(256, 1): full VGPR budget (live set ~170 VGPRs) so the
// accumulators never spill to scratch; 2 waves/SIMD + async pipeline is
// sufficient latency hiding for a bandwidth-bound GEMM.
// Requires: M % 256 == 0, NT % 4 == 0 (all GEMMs here padded to satisfy).
// Grid: (M/256, NT/4).
// ---------------------------------------------------------------------------
template <int ACT>   // 0 = none, 1 = silu
__global__ __launch_bounds__(256, 1)
void gemm_bf16_v2(const unsigned short* __restrict__ Apack,
                  const unsigned short* __restrict__ Bpack,
                  const float* __restrict__ bias,
                  float* __restrict__ out,
                  int Ncols, int KT, int NT) {
    __shared__ __align__(32) unsigned short sB[2][4 * 32 * 16];  // 2 x 4KB
    const v16bf* Ap = (const v16bf*)Apack;

    int tid  = threadIdx.x;
    int wave = tid >> 5;
    int lane = tid & 31;
    int mt0  = blockIdx.x * 16 + wave * 2;
    int nt0  = blockIdx.y * 4;

    // per-thread async-copy slot: 256 threads x 16B = one 4KB B batch
    int cj    = tid >> 6;          // tile 0..3
    int clane = (tid >> 1) & 31;
    int chalf = tid & 1;
    size_t srcBase = ((size_t)(nt0 + cj) * 32 + clane) * 16 + (size_t)chalf * 8; // ushorts
    size_t kStep   = (size_t)NT * 32 * 16;                                       // ushorts / kt
    unsigned ldsOff[2];
    ldsOff[0] = (unsigned)(size_t)&sB[0][(cj * 32 + clane) * 16 + chalf * 8];
    ldsOff[1] = (unsigned)(size_t)&sB[1][(cj * 32 + clane) * 16 + chalf * 8];

    auto issueB = [&](int kt, int buf) {
        const unsigned short* g = Bpack + srcBase + (size_t)kt * kStep;
        asm volatile("global_load_async_to_lds_b128 %0, %1, off"
                     :: "v"(ldsOff[buf]), "v"((unsigned long long)(uintptr_t)g)
                     : "memory");
    };

    v8f acc[2][4] = {};
    size_t aRow0 = (size_t)mt0 * KT;
    size_t aRow1 = (size_t)(mt0 + 1) * KT;

    issueB(0, 0);
    v16bf a0 = Ap[aRow0 * 32 + lane];
    v16bf a1 = Ap[aRow1 * 32 + lane];

    for (int kt = 0; kt < KT; ++kt) {
        int nxt = (kt + 1 < KT) ? kt + 1 : KT - 1;   // clamped: branch-free tail
        issueB(nxt, (kt + 1) & 1);
        v16bf an0 = Ap[(aRow0 + nxt) * 32 + lane];   // register-pipelined A
        v16bf an1 = Ap[(aRow1 + nxt) * 32 + lane];
        __builtin_prefetch((const char*)&Ap[(aRow0 + ((kt + 2 < KT) ? kt + 2 : KT - 1)) * 32 + lane], 0, 1);

        asm volatile("s_wait_asynccnt 0x1" ::: "memory");  // batch kt resident
        __syncthreads();
        const v16bf* bs = (const v16bf*)&sB[kt & 1][0];
        v16bf b0 = bs[0 * 32 + lane];
        v16bf b1 = bs[1 * 32 + lane];
        v16bf b2 = bs[2 * 32 + lane];
        v16bf b3 = bs[3 * 32 + lane];
        __syncthreads();   // all waves hold B in regs; LDS buf reusable

        acc[0][0] = wmma_bf16(a0, b0, acc[0][0]);
        acc[0][1] = wmma_bf16(a0, b1, acc[0][1]);
        acc[0][2] = wmma_bf16(a0, b2, acc[0][2]);
        acc[0][3] = wmma_bf16(a0, b3, acc[0][3]);
        acc[1][0] = wmma_bf16(a1, b0, acc[1][0]);
        acc[1][1] = wmma_bf16(a1, b1, acc[1][1]);
        acc[1][2] = wmma_bf16(a1, b2, acc[1][2]);
        acc[1][3] = wmma_bf16(a1, b3, acc[1][3]);

        a0 = an0; a1 = an1;
    }

    // D layout: lane = g*16 + n ; VGPR r holds M = r + 8*g, N = n.
    int g = lane >> 4, n = lane & 15;
    #pragma unroll
    for (int j = 0; j < 4; ++j) {
        int col = (nt0 + j) * 16 + n;
        float bv = bias ? bias[col] : 0.f;
        #pragma unroll
        for (int mi = 0; mi < 2; ++mi) {
            #pragma unroll
            for (int r = 0; r < 8; ++r) {
                int row = (mt0 + mi) * 16 + r + 8 * g;
                float v = acc[mi][j][r] + bv;
                if (ACT == 1) v = silu_f(v);
                out[(size_t)row * Ncols + col] = v;
            }
        }
    }
}

// ---------------------------------------------------------------------------
// Depthwise causal conv (K=4) over L, + bias + SiLU. xm = xz[:, :512].
// ---------------------------------------------------------------------------
__global__ void conv_silu_k(const float* __restrict__ xz,
                            const float* __restrict__ cw,
                            const float* __restrict__ cb,
                            float* __restrict__ xconv) {
    int idx = blockIdx.x * blockDim.x + threadIdx.x;
    if (idx >= MM * DI) return;
    int d = idx & (DI - 1);
    int m = idx >> 9;           // row in B*L
    int l = m & (LL - 1);
    float acc = cb[d];
    #pragma unroll
    for (int k = 0; k < KCONV; ++k) {
        int t = l + k - (KCONV - 1);
        if (t >= 0) acc += cw[d * KCONV + k] * xz[(size_t)(m + k - (KCONV - 1)) * (2 * DI) + d];
    }
    xconv[(size_t)m * DI + d] = silu_f(acc);
}

// ---------------------------------------------------------------------------
// dt = softplus(dbc[:, :16] @ dt_proj_w + dt_proj_b)   (K=16 -> plain VALU)
// ---------------------------------------------------------------------------
__global__ void dt_softplus_k(const float* __restrict__ dbc,
                              const float* __restrict__ dtw,
                              const float* __restrict__ dtb,
                              float* __restrict__ dt) {
    int idx = blockIdx.x * blockDim.x + threadIdx.x;
    if (idx >= MM * DI) return;
    int d = idx & (DI - 1);
    int m = idx >> 9;
    float acc = dtb[d];
    #pragma unroll
    for (int k = 0; k < DTR; ++k)
        acc += dbc[(size_t)m * DBCN + k] * dtw[k * DI + d];
    float sp = (acc > 20.f) ? acc : log1pf(__expf(acc));
    dt[(size_t)m * DI + d] = sp;
}

// ---------------------------------------------------------------------------
// Selective scan: one wave per (b, channel d); lane = state index n (N=32).
// h[n] lives in one register per lane; y_t = wave butterfly reduction.
// Epilogue fused: + xm*D_skip, * silu(z).
// ---------------------------------------------------------------------------
__global__ void scan_k(const float* __restrict__ xconv,
                       const float* __restrict__ dtbuf,
                       const float* __restrict__ dbc,
                       const float* __restrict__ xz,
                       const float* __restrict__ A_log,
                       const float* __restrict__ D_skip,
                       float* __restrict__ ybuf) {
    int wid  = (blockIdx.x * blockDim.x + threadIdx.x) >> 5;
    int lane = threadIdx.x & 31;
    if (wid >= BB * DI) return;
    int b = wid >> 9;
    int d = wid & (DI - 1);
    float a = -__expf(A_log[d * NN + lane]);
    float dskip = D_skip[d];
    float h = 0.f;
    size_t base = (size_t)b * LL;
    for (int t = 0; t < LL; ++t) {
        size_t m = base + t;
        float xt  = xconv[m * DI + d];
        float dtv = dtbuf[m * DI + d];
        float Bt  = dbc[m * DBCN + DTR + lane];
        float Ct  = dbc[m * DBCN + DTR + NN + lane];
        float dA  = __expf(dtv * a);
        h = dA * h + (dtv * xt) * Bt;
        float p = h * Ct;
        #pragma unroll
        for (int o = 16; o >= 1; o >>= 1) p += __shfl_xor(p, o, 32);
        if (lane == 0) {
            float z = xz[m * (2 * DI) + DI + d];
            ybuf[m * DI + d] = (p + xt * dskip) * silu_f(z);
        }
    }
}

// ---------------------------------------------------------------------------
// Final gate mix: out = motion + g*ssm + (1-g)*constraints, g = sigmoid(gpre)
// ---------------------------------------------------------------------------
__global__ void final_mix_k(const float* __restrict__ motion,
                            const float* __restrict__ ssm,
                            const float* __restrict__ constr,
                            const float* __restrict__ gpre,
                            float* __restrict__ out) {
    int idx = blockIdx.x * blockDim.x + threadIdx.x;
    if (idx >= MM * DD) return;
    float g = sigmoid_f(gpre[idx]);
    out[idx] = motion[idx] + g * ssm[idx] + (1.f - g) * constr[idx];
}

// ---------------------------------------------------------------------------
// launch
// ---------------------------------------------------------------------------
extern "C" void kernel_launch(void* const* d_in, const int* in_sizes, int n_in,
                              void* d_out, int out_size, void* d_ws, size_t ws_size,
                              hipStream_t stream) {
    const float* motion   = (const float*)d_in[0];
    const float* physics  = (const float*)d_in[1];
    const float* norm_w   = (const float*)d_in[2];
    const float* norm_b   = (const float*)d_in[3];
    const float* in_w     = (const float*)d_in[4];
    const float* in_b     = (const float*)d_in[5];
    const float* conv_w   = (const float*)d_in[6];
    const float* conv_b   = (const float*)d_in[7];
    const float* xproj_w  = (const float*)d_in[8];
    const float* dtproj_w = (const float*)d_in[9];
    const float* dtproj_b = (const float*)d_in[10];
    const float* A_log    = (const float*)d_in[11];
    const float* D_skip   = (const float*)d_in[12];
    const float* out_w    = (const float*)d_in[13];
    const float* out_b    = (const float*)d_in[14];
    const float* pe1_w    = (const float*)d_in[15];
    const float* pe1_b    = (const float*)d_in[16];
    const float* pe2_w    = (const float*)d_in[17];
    const float* pe2_b    = (const float*)d_in[18];
    const float* cp_w     = (const float*)d_in[19];
    const float* cp_b     = (const float*)d_in[20];
    const float* gate_w   = (const float*)d_in[21];
    const float* gate_b   = (const float*)d_in[22];
    float* outp = (float*)d_out;

    // ---- carve workspace (256B aligned) ----
    char* base = (char*)d_ws;
    size_t off = 0;
    auto carve = [&](size_t bytes) -> void* {
        void* p = base + off;
        off = (off + bytes + 255) & ~(size_t)255;
        return p;
    };
    float*          xn     = (float*)         carve((size_t)MM * DD * 4);
    unsigned short* xnP    = (unsigned short*)carve((size_t)MM * DD * 2);
    unsigned short* wInP   = (unsigned short*)carve((size_t)DD * 1024 * 2);
    float*          xz     = (float*)         carve((size_t)MM * 1024 * 4);
    float*          xconv  = (float*)         carve((size_t)MM * DI * 4);
    unsigned short* xcP    = (unsigned short*)carve((size_t)MM * DI * 2);
    unsigned short* wXP    = (unsigned short*)carve((size_t)DI * DBCN * 2);
    float*          dbc    = (float*)         carve((size_t)MM * DBCN * 4);
    float*          dtb    = (float*)         carve((size_t)MM * DI * 4);
    float*          ybuf   = (float*)         carve((size_t)MM * DI * 4);
    unsigned short* yP     = (unsigned short*)carve((size_t)MM * DI * 2);
    unsigned short* wOutP  = (unsigned short*)carve((size_t)DI * DD * 2);
    float*          ssm    = (float*)         carve((size_t)MM * DD * 4);
    unsigned short* physP  = (unsigned short*)carve((size_t)MM * PP * 2);
    unsigned short* wPe1P  = (unsigned short*)carve((size_t)PP * DD * 2);
    float*          ptmp   = (float*)         carve((size_t)MM * DD * 4);
    unsigned short* ptmpP  = (unsigned short*)carve((size_t)MM * DD * 2);
    unsigned short* wPe2P  = (unsigned short*)carve((size_t)DD * DD * 2);
    float*          pemb   = (float*)         carve((size_t)MM * DD * 4);
    unsigned short* wCpP   = (unsigned short*)carve((size_t)PP * DD * 2);
    float*          constr = (float*)         carve((size_t)MM * DD * 4);
    unsigned short* catP   = (unsigned short*)carve((size_t)MM * (2 * DD) * 2);
    unsigned short* wGateP = (unsigned short*)carve((size_t)(2 * DD) * DD * 2);
    float*          gpre   = (float*)         carve((size_t)MM * DD * 4);
    (void)ws_size; (void)n_in; (void)in_sizes; (void)out_size;

    const int MT = MM / 16;   // 1024
    auto packAGrid = [](int KT) { return (MT * KT * 32 + 255) / 256; };
    auto packWGrid = [](int KT, int NT) { return (KT * NT * 32 + 255) / 256; };
    auto gemmGrid  = [](int NT) { return dim3(MM / 256, NT / 4); };

    // 1) layernorm
    layernorm_k<<<MM, DD, 0, stream>>>(motion, norm_w, norm_b, xn);

    // 2) in_proj: (M,256)x(256,1024) -> xz
    pack_a_k<<<packAGrid(8), 256, 0, stream>>>(xn, xnP, MT, 8, DD, 0, 8);
    pack_w_k<<<packWGrid(8, 64), 256, 0, stream>>>(in_w, wInP, 8, 64, 1024);
    gemm_bf16_v2<0><<<gemmGrid(64), 256, 0, stream>>>(xnP, wInP, in_b, xz, 1024, 8, 64);

    // 3) depthwise conv + silu
    conv_silu_k<<<(MM * DI + 255) / 256, 256, 0, stream>>>(xz, conv_w, conv_b, xconv);

    // 4) x_proj: (M,512)x(512,80->128 padded) -> dbc
    pack_a_k<<<packAGrid(16), 256, 0, stream>>>(xconv, xcP, MT, 16, DI, 0, 16);
    pack_w_k<<<packWGrid(16, 8), 256, 0, stream>>>(xproj_w, wXP, 16, 8, 80);
    gemm_bf16_v2<0><<<gemmGrid(8), 256, 0, stream>>>(xcP, wXP, nullptr, dbc, DBCN, 16, 8);

    // 5) dt projection + softplus
    dt_softplus_k<<<(MM * DI + 255) / 256, 256, 0, stream>>>(dbc, dtproj_w, dtproj_b, dtb);

    // 6) selective scan (wave per channel, lane = state) + D_skip + z-gate
    scan_k<<<(BB * DI * 32 + 255) / 256, 256, 0, stream>>>(xconv, dtb, dbc, xz, A_log, D_skip, ybuf);

    // 7) out_proj: (M,512)x(512,256) -> ssm_out
    pack_a_k<<<packAGrid(16), 256, 0, stream>>>(ybuf, yP, MT, 16, DI, 0, 16);
    pack_w_k<<<packWGrid(16, 16), 256, 0, stream>>>(out_w, wOutP, 16, 16, DD);
    gemm_bf16_v2<0><<<gemmGrid(16), 256, 0, stream>>>(yP, wOutP, out_b, ssm, DD, 16, 16);

    // 8) physics embed: silu(phys@pe1)@pe2 ; constraints = phys@cp
    pack_a_k<<<packAGrid(2), 256, 0, stream>>>(physics, physP, MT, 2, PP, 0, 2);
    pack_w_k<<<packWGrid(2, 16), 256, 0, stream>>>(pe1_w, wPe1P, 2, 16, DD);
    gemm_bf16_v2<1><<<gemmGrid(16), 256, 0, stream>>>(physP, wPe1P, pe1_b, ptmp, DD, 2, 16);
    pack_a_k<<<packAGrid(8), 256, 0, stream>>>(ptmp, ptmpP, MT, 8, DD, 0, 8);
    pack_w_k<<<packWGrid(8, 16), 256, 0, stream>>>(pe2_w, wPe2P, 8, 16, DD);
    gemm_bf16_v2<0><<<gemmGrid(16), 256, 0, stream>>>(ptmpP, wPe2P, pe2_b, pemb, DD, 8, 16);
    pack_w_k<<<packWGrid(2, 16), 256, 0, stream>>>(cp_w, wCpP, 2, 16, DD);
    gemm_bf16_v2<0><<<gemmGrid(16), 256, 0, stream>>>(physP, wCpP, cp_b, constr, DD, 2, 16);

    // 9) gate: concat(ssm, pemb) (K=512) @ gate_w -> gpre
    pack_a_k<<<packAGrid(8), 256, 0, stream>>>(ssm,  catP, MT, 8, DD, 0, 16);
    pack_a_k<<<packAGrid(8), 256, 0, stream>>>(pemb, catP, MT, 8, DD, 8, 16);
    pack_w_k<<<packWGrid(16, 16), 256, 0, stream>>>(gate_w, wGateP, 16, 16, DD);
    gemm_bf16_v2<0><<<gemmGrid(16), 256, 0, stream>>>(catP, wGateP, gate_b, gpre, DD, 16, 16);

    // 10) final mix
    final_mix_k<<<(MM * DD + 255) / 256, 256, 0, stream>>>(motion, ssm, constr, gpre, outp);
}
